// AdaAtt_attention_64476049047929
// MI455X (gfx1250) — matrix operand order repaired and checked by reading, not compiled, not run on hardware
//
#include <hip/hip_runtime.h>
#include <hip/hip_bf16.h>

// ---------------------------------------------------------------------------
// AdaAtt attention, fused for MI455X (gfx1250, wave32, WMMA bf16 16x16x32).
// B=2048, A=196, D=512.
// ---------------------------------------------------------------------------

#define DDIM   512
#define AREG   196
#define BROWS  2048
#define MTILE  64            // GEMM rows per workgroup
#define KCHUNK 256           // K staged in LDS per pass
#define LDA    272           // KCHUNK + 16 bf16 pad (bank spread)

typedef __bf16 v16bf __attribute__((ext_vector_type(16)));
typedef __bf16 v8bf  __attribute__((ext_vector_type(8)));
typedef float  v8f   __attribute__((ext_vector_type(8)));

__device__ __forceinline__ unsigned short f2bf(float f) {
  unsigned int u = __float_as_uint(f);
  u += 0x7FFFu + ((u >> 16) & 1u);            // round-to-nearest-even
  return (unsigned short)(u >> 16);
}

// ---------------------------------------------------------------------------
// Weight convert: f32 -> bf16 (weights stay L2-resident afterwards)
// ---------------------------------------------------------------------------
__global__ __launch_bounds__(256) void f32_to_bf16_kernel(
    const float* __restrict__ src, unsigned short* __restrict__ dst, int n4) {
  int i = blockIdx.x * 256 + threadIdx.x;
  if (i < n4) {
    float4 v = ((const float4*)src)[i];
    ushort4 h;
    h.x = f2bf(v.x); h.y = f2bf(v.y); h.z = f2bf(v.z); h.w = f2bf(v.w);
    ((ushort4*)dst)[i] = h;
  }
}

// ---------------------------------------------------------------------------
// GEMM core helpers (C[64x512] = A[64x512] @ W.T, bf16 WMMA, f32 accum)
// ---------------------------------------------------------------------------
__device__ __forceinline__ void load_chunk(const float* __restrict__ A,
                                           int rowBase, int kc,
                                           unsigned short* ldsA, int tid) {
  // 64 rows x 256 cols (f32 -> bf16 into LDS), 256 threads, 16 iters each
  #pragma unroll 1
  for (int idx = tid; idx < MTILE * (KCHUNK / 4); idx += 256) {
    int row = idx >> 6;
    int c4  = (idx & 63) << 2;
    float4 v = *(const float4*)(A + (size_t)(rowBase + row) * DDIM + kc + c4);
    ushort4 h;
    h.x = f2bf(v.x); h.y = f2bf(v.y); h.z = f2bf(v.z); h.w = f2bf(v.w);
    *(ushort4*)(ldsA + row * LDA + c4) = h;
  }
}

// One K-chunk of WMMAs. Wave w owns N-tiles [4w,4w+4), all 4 M-subtiles.
__device__ __forceinline__ void wmma_chunk(const unsigned short* ldsA,
                                           const unsigned short* __restrict__ Wbf,
                                           int kc, int w, int lane15, int kh,
                                           v8f acc[4][4]) {
  #pragma unroll 1
  for (int k0 = 0; k0 < KCHUNK; k0 += 32) {
    v16bf af[4];
    #pragma unroll
    for (int mi = 0; mi < 4; ++mi) {
      // A-layout 16x32 bf16: lane holds K = [k0+16*kh, +16) contiguous
      const unsigned short* p = ldsA + (mi * 16 + lane15) * LDA + k0 + 16 * kh;
      v8bf a0 = *(const v8bf*)p;
      v8bf a1 = *(const v8bf*)(p + 8);
      af[mi] = __builtin_shufflevector(a0, a1,
               0, 1, 2, 3, 4, 5, 6, 7, 8, 9, 10, 11, 12, 13, 14, 15);
    }
    #pragma unroll
    for (int j = 0; j < 4; ++j) {
      // B-layout 32x16: lane n holds W[n][k0+16*kh .. +15] contiguous
      const int ncol = (w * 4 + j) * 16 + lane15;
      v16bf bf = *(const v16bf*)(Wbf + (size_t)ncol * DDIM + kc + k0 + 16 * kh);
      #pragma unroll
      for (int mi = 0; mi < 4; ++mi) {
        acc[mi][j] = __builtin_amdgcn_wmma_f32_16x16x32_bf16(
            false, af[mi], false, bf, (short)0, acc[mi][j], false, false);
      }
    }
  }
}

// ---------------------------------------------------------------------------
// Generic GEMM + bias + activation (ACT: 0=none, 1=relu, 2=tanh)
// ---------------------------------------------------------------------------
template <int ACT>
__global__ __launch_bounds__(256) void gemm512_kernel(
    const float* __restrict__ A, const unsigned short* __restrict__ Wbf,
    const float* __restrict__ bias, float* __restrict__ C) {
  __shared__ unsigned short ldsA[MTILE * LDA];
  const int tid = threadIdx.x;
  const int w = tid >> 5, lane = tid & 31, lane15 = lane & 15, kh = lane >> 4;
  const int rowBase = blockIdx.x * MTILE;

  v8f acc[4][4];
  #pragma unroll
  for (int mi = 0; mi < 4; ++mi) {
    #pragma unroll
    for (int j = 0; j < 4; ++j) {
      #pragma unroll
      for (int e = 0; e < 8; ++e) acc[mi][j][e] = 0.0f;
    }
  }

  for (int kc = 0; kc < DDIM; kc += KCHUNK) {
    __syncthreads();
    load_chunk(A, rowBase, kc, ldsA, tid);
    __syncthreads();
    wmma_chunk(ldsA, Wbf, kc, w, lane15, kh, acc);
  }

  // C-tile layout: VGPR e -> row (e or e+8 by lane half), col = lane15
  #pragma unroll
  for (int j = 0; j < 4; ++j) {
    const int col = (w * 4 + j) * 16 + lane15;
    const float bv = bias[col];
    #pragma unroll
    for (int mi = 0; mi < 4; ++mi) {
      #pragma unroll
      for (int e = 0; e < 8; ++e) {
        const int m = mi * 16 + e + kh * 8;
        float v = acc[mi][j][e] + bv;
        if (ACT == 1) v = fmaxf(v, 0.0f);
        if (ACT == 2) v = tanhf(v);
        C[(size_t)(rowBase + m) * DDIM + col] = v;
      }
    }
  }
}

// ---------------------------------------------------------------------------
// Fused big kernel: rows r=(b*196+a) of conv_feat_embed.
// logit[b, 1+a] = sum_n tanh(gemm + cfe_b[n] + hoe[b][n]) * alpha[n]
// (the 822MB cfe tensor is never written to memory)
// ---------------------------------------------------------------------------
__global__ __launch_bounds__(256) void gemm512_attn_kernel(
    const float* __restrict__ A, const unsigned short* __restrict__ Wbf,
    const float* __restrict__ bias, const float* __restrict__ hoe,
    const float* __restrict__ alphaW, const float* __restrict__ alphaB,
    float* __restrict__ logits) {
  __shared__ unsigned short ldsA[MTILE * LDA];
  __shared__ float sm_logit[MTILE];
  const int tid = threadIdx.x;
  const int w = tid >> 5, lane = tid & 31, lane15 = lane & 15, kh = lane >> 4;
  const int rowBase = blockIdx.x * MTILE;

  if (tid < MTILE) sm_logit[tid] = 0.0f;

  v8f acc[4][4];
  #pragma unroll
  for (int mi = 0; mi < 4; ++mi) {
    #pragma unroll
    for (int j = 0; j < 4; ++j) {
      #pragma unroll
      for (int e = 0; e < 8; ++e) acc[mi][j][e] = 0.0f;
    }
  }

  for (int kc = 0; kc < DDIM; kc += KCHUNK) {
    __syncthreads();   // first pass also covers sm_logit init
    load_chunk(A, rowBase, kc, ldsA, tid);
    __syncthreads();
    wmma_chunk(ldsA, Wbf, kc, w, lane15, kh, acc);
  }

  #pragma unroll
  for (int mi = 0; mi < 4; ++mi) {
    #pragma unroll
    for (int e = 0; e < 8; ++e) {
      const int m = mi * 16 + e + kh * 8;
      const int r = rowBase + m;
      const int b = r / AREG;
      float part = 0.0f;
      #pragma unroll
      for (int j = 0; j < 4; ++j) {
        const int col = (w * 4 + j) * 16 + lane15;
        float v = acc[mi][j][e] + bias[col] + hoe[(size_t)b * DDIM + col];
        part += tanhf(v) * alphaW[col];
      }
      // reduce over the 16 lanes sharing this output row
      #pragma unroll
      for (int off = 1; off < 16; off <<= 1)
        part += __shfl_xor(part, off, 32);
      if (lane15 == 0) atomicAdd(&sm_logit[m], part);   // ds_add_f32
    }
  }
  __syncthreads();
  if (tid < MTILE) {
    const int r = rowBase + tid;
    logits[(size_t)(r / AREG) * (AREG + 1) + 1 + (r % AREG)] =
        sm_logit[tid] + alphaB[0];
  }
}

// ---------------------------------------------------------------------------
// logits[:,0] = dot(tanh(fre + hoe), alpha) + alpha_b
// ---------------------------------------------------------------------------
__global__ __launch_bounds__(256) void logit0_kernel(
    const float* __restrict__ fre, const float* __restrict__ hoe,
    const float* __restrict__ alphaW, const float* __restrict__ alphaB,
    float* __restrict__ logits) {
  const int b = blockIdx.x, tid = threadIdx.x;
  float p = 0.0f;
  for (int d = tid; d < DDIM; d += 256)
    p += tanhf(fre[(size_t)b * DDIM + d] + hoe[(size_t)b * DDIM + d]) * alphaW[d];
  __shared__ float red[256];
  red[tid] = p;
  __syncthreads();
  for (int s = 128; s > 0; s >>= 1) {
    if (tid < s) red[tid] += red[tid + s];
    __syncthreads();
  }
  if (tid == 0) logits[(size_t)b * (AREG + 1)] = red[0] + alphaB[0];
}

// ---------------------------------------------------------------------------
// Softmax over 197 logits, visAtt = sum_a PI[a]*img_all[a], beta-blend output.
// Streams conv_feat (822 MB) coalesced, float2 per lane.
// ---------------------------------------------------------------------------
__global__ __launch_bounds__(256) void attend_kernel(
    const float* __restrict__ logits, const float* __restrict__ fr,
    const float* __restrict__ conv_feat, float* __restrict__ out) {
  const int b = blockIdx.x, tid = threadIdx.x;
  __shared__ float pi[AREG + 1];
  __shared__ float red[256];

  float v = (tid <= AREG) ? logits[(size_t)b * (AREG + 1) + tid] : -3.4e38f;
  red[tid] = v;
  __syncthreads();
  for (int s = 128; s > 0; s >>= 1) {
    if (tid < s) red[tid] = fmaxf(red[tid], red[tid + s]);
    __syncthreads();
  }
  const float mx = red[0];
  __syncthreads();
  float e = (tid <= AREG) ? __expf(v - mx) : 0.0f;
  red[tid] = e;
  __syncthreads();
  for (int s = 128; s > 0; s >>= 1) {
    if (tid < s) red[tid] += red[tid + s];
    __syncthreads();
  }
  const float inv = 1.0f / red[0];
  if (tid <= AREG) pi[tid] = e * inv;
  __syncthreads();

  const int d = tid * 2;
  const float2 frv = *(const float2*)(fr + (size_t)b * DDIM + d);
  float ax = pi[0] * frv.x, ay = pi[0] * frv.y;
  const float* cf = conv_feat + (size_t)b * AREG * DDIM + d;
  #pragma unroll 4
  for (int a = 0; a < AREG; ++a) {
    const float2 c = *(const float2*)(cf + (size_t)a * DDIM);
    const float wgt = pi[a + 1];
    ax += wgt * c.x;
    ay += wgt * c.y;
  }
  const float beta = pi[AREG];
  float2 o;
  o.x = beta * frv.x + (1.0f - beta) * ax;
  o.y = beta * frv.y + (1.0f - beta) * ay;
  *(float2*)(out + (size_t)b * DDIM + d) = o;
}

// ---------------------------------------------------------------------------
extern "C" void kernel_launch(void* const* d_in, const int* in_sizes, int n_in,
                              void* d_out, int out_size, void* d_ws, size_t ws_size,
                              hipStream_t stream) {
  const float* h_out       = (const float*)d_in[0];
  const float* fake_region = (const float*)d_in[1];
  const float* conv_feat   = (const float*)d_in[2];
  const float* conv_embed  = (const float*)d_in[3];
  const float* fr_W  = (const float*)d_in[4];
  const float* fr_b  = (const float*)d_in[5];
  const float* fre_W = (const float*)d_in[6];
  const float* fre_b = (const float*)d_in[7];
  const float* ho_W  = (const float*)d_in[8];
  const float* ho_b  = (const float*)d_in[9];
  const float* hoe_W = (const float*)d_in[10];
  const float* hoe_b = (const float*)d_in[11];
  const float* alpha_W = (const float*)d_in[12];
  const float* alpha_b = (const float*)d_in[13];
  const float* cfe_W = (const float*)d_in[14];
  const float* cfe_b = (const float*)d_in[15];
  float* out = (float*)d_out;

  char* ws = (char*)d_ws;
  size_t o = 0;
  auto take = [&](size_t bytes) -> char* {
    char* p = ws + o;
    o += (bytes + 255) & ~(size_t)255;
    return p;
  };
  unsigned short* WfrB  = (unsigned short*)take((size_t)DDIM * DDIM * 2);
  unsigned short* WfreB = (unsigned short*)take((size_t)DDIM * DDIM * 2);
  unsigned short* WhoB  = (unsigned short*)take((size_t)DDIM * DDIM * 2);
  unsigned short* WhoeB = (unsigned short*)take((size_t)DDIM * DDIM * 2);
  unsigned short* WcfeB = (unsigned short*)take((size_t)DDIM * DDIM * 2);
  float* fr     = (float*)take((size_t)BROWS * DDIM * 4);
  float* hol    = (float*)take((size_t)BROWS * DDIM * 4);
  float* fre    = (float*)take((size_t)BROWS * DDIM * 4);
  float* hoe    = (float*)take((size_t)BROWS * DDIM * 4);
  float* logits = (float*)take((size_t)BROWS * (AREG + 1) * 4);

  const int n4 = DDIM * DDIM / 4;           // 65536 float4s per matrix
  const int cgrid = n4 / 256;
  f32_to_bf16_kernel<<<cgrid, 256, 0, stream>>>(fr_W,  WfrB,  n4);
  f32_to_bf16_kernel<<<cgrid, 256, 0, stream>>>(fre_W, WfreB, n4);
  f32_to_bf16_kernel<<<cgrid, 256, 0, stream>>>(ho_W,  WhoB,  n4);
  f32_to_bf16_kernel<<<cgrid, 256, 0, stream>>>(hoe_W, WhoeB, n4);
  f32_to_bf16_kernel<<<cgrid, 256, 0, stream>>>(cfe_W, WcfeB, n4);

  // fr = relu(fake_region @ fr_W.T + fr_b); hol = tanh(h_out @ ho_W.T + ho_b)
  gemm512_kernel<1><<<BROWS / MTILE, 256, 0, stream>>>(fake_region, WfrB, fr_b, fr);
  gemm512_kernel<2><<<BROWS / MTILE, 256, 0, stream>>>(h_out, WhoB, ho_b, hol);
  // fre = fr @ fre_W.T + fre_b; hoe = hol @ hoe_W.T + hoe_b
  gemm512_kernel<0><<<BROWS / MTILE, 256, 0, stream>>>(fr, WfreB, fre_b, fre);
  gemm512_kernel<0><<<BROWS / MTILE, 256, 0, stream>>>(hol, WhoeB, hoe_b, hoe);

  // big fused GEMM -> logits[:,1:]
  gemm512_attn_kernel<<<(BROWS * AREG) / MTILE, 256, 0, stream>>>(
      conv_embed, WcfeB, cfe_b, hoe, alpha_W, alpha_b, logits);
  // logits[:,0]
  logit0_kernel<<<BROWS, 256, 0, stream>>>(fre, hoe, alpha_W, alpha_b, logits);
  // softmax + visAtt + blend
  attend_kernel<<<BROWS, 256, 0, stream>>>(logits, fr, conv_feat, out);
}
